// NeuralODE_54717883351355
// MI455X (gfx1250) — compile-verified, matrix operand order checked by reading, not run on hardware
//
#include <hip/hip_runtime.h>

#define B_TOT 4096
#define SEQ   256
#define NDOF  7
#define SD    14
#define HID   256
#define INDIM 21
#define KPAD  32
#define ROWS  16
#define DT_F  0.002f

typedef __attribute__((ext_vector_type(16))) __bf16 v16bf;
typedef __attribute__((ext_vector_type(8)))  __bf16 v8bf;
typedef __attribute__((ext_vector_type(8)))  float  v8f;

union FragU { v16bf v; v8bf h[2]; __bf16 e[16]; };

// Fast silu: v_exp_f32 + v_add + v_rcp_f32 + v_mul (no IEEE div expansion).
__device__ __forceinline__ float silu_f(float x) {
    return x * __builtin_amdgcn_rcpf(1.0f + __expf(-x));
}

// A-matrix fragment (16x32 bf16) from row-major LDS buffer.
// ISA layout: lanes 0-15 row M=lane, halves 0-7 = K 0..7, halves 8-15 = K 16..23;
// lanes 16-31 row M=lane-16, halves 0-7 = K 8..15, halves 8-15 = K 24..31.
__device__ __forceinline__ v16bf load_a_frag(const __bf16* base, int ldk, int kbase, int lane) {
    int m  = lane & 15;
    int ko = kbase + ((lane >> 4) << 3);
    const __bf16* p = base + m * ldk + ko;
    FragU f;
    f.h[0] = *(const v8bf*)(p);        // K ko..ko+7   (16B-aligned ds_load_b128)
    f.h[1] = *(const v8bf*)(p + 16);   // K ko+16..ko+23
    return f.v;
}

// B-matrix fragment (32x16 bf16) built from row-major f32 global weights (done once).
__device__ __forceinline__ v16bf make_b_frag(const float* W, int ncols, int kbase, int ntile,
                                             int lane, int kmax, int nmax) {
    FragU f;
    int n  = ntile * 16 + (lane & 15);
    int ko = kbase + ((lane >> 4) << 3);
    #pragma unroll
    for (int i = 0; i < 16; ++i) {
        int k = ko + (i & 7) + ((i >> 3) << 4);
        float val = (k < kmax && n < nmax) ? W[(size_t)k * ncols + n] : 0.0f;
        f.e[i] = (__bf16)val;
    }
    return f.v;
}

__global__ __launch_bounds__(256) void node_verlet_kernel(
    const float* __restrict__ init_state, const float* __restrict__ torq,
    const float* __restrict__ W1, const float* __restrict__ b1,
    const float* __restrict__ W2, const float* __restrict__ b2,
    const float* __restrict__ W3, const float* __restrict__ b3,
    float* __restrict__ out)
{
    __shared__ __align__(16) __bf16 xs [ROWS * KPAD];   // MLP input, padded 21->32
    __shared__ __align__(16) __bf16 h1s[ROWS * HID];    // layer-1 activations (bf16)
    __shared__ __align__(16) __bf16 h2s[ROWS * HID];    // layer-2 activations (bf16)
    __shared__ float part[8 * ROWS * 16];               // layer-3 per-wave partials

    const int tid  = threadIdx.x;
    const int lane = tid & 31;
    const int wave = tid >> 5;
    const int r0   = blockIdx.x * ROWS;

    // Per-thread state ownership: threads 0..111 each own one (row m, dof c).
    const bool owner = (tid < ROWS * NDOF);
    const int  om    = tid / NDOF;       // row within tile (valid if owner)
    const int  oc    = tid % NDOF;       // dof index       (valid if owner)

    // ---- Preload all weights into WMMA B-fragment registers (once) ----
    v16bf w1f[2], w2f[2][8], w3f;
    float b1v[2], b2v[2];
    #pragma unroll
    for (int t = 0; t < 2; ++t) {
        int nt = wave * 2 + t;
        w1f[t] = make_b_frag(W1, HID, 0, nt, lane, INDIM, HID);
        b1v[t] = b1[nt * 16 + (lane & 15)];
        b2v[t] = b2[nt * 16 + (lane & 15)];
        #pragma unroll
        for (int kc = 0; kc < 8; ++kc)
            w2f[t][kc] = make_b_frag(W2, HID, kc * 32, nt, lane, HID, HID);
    }
    w3f = make_b_frag(W3, NDOF, wave * 32, 0, lane, HID, NDOF);

    const float b3v = owner ? b3[oc] : 0.0f;

    // Store silu(acc) into bf16 LDS buffer. C/D layout: lane -> N=lane&15,
    // VGPR v -> M = v + (lane>=16 ? 8 : 0).
    auto store_silu = [&](__bf16* h, v8f c, int ntile) {
        int n  = ntile * 16 + (lane & 15);
        int mh = (lane >> 4) << 3;
        #pragma unroll
        for (int v = 0; v < 8; ++v)
            h[(v + mh) * HID + n] = (__bf16)silu_f(c[v]);
    };

    float anew = 0.0f;   // layer-3 result for the owner thread's (m,c)

    // One MLP eval: xs -> anew. Caller must __syncthreads() after xs is written.
    auto accel_mlp = [&]() {
        // layer 1: x(16x32) @ W1(32x256) + b1, silu
        {
            v16bf xa = load_a_frag(xs, KPAD, 0, lane);
            #pragma unroll
            for (int t = 0; t < 2; ++t) {
                v8f c;
                #pragma unroll
                for (int v = 0; v < 8; ++v) c[v] = b1v[t];
                c = __builtin_amdgcn_wmma_f32_16x16x32_bf16(
                        false, xa, false, w1f[t], (short)0, c, false, false);
                store_silu(h1s, c, wave * 2 + t);
            }
        }
        __syncthreads();
        // layer 2: h1(16x256) @ W2(256x256) + b2, silu (two interleaved acc chains)
        {
            v8f c0, c1;
            #pragma unroll
            for (int v = 0; v < 8; ++v) { c0[v] = b2v[0]; c1[v] = b2v[1]; }
            #pragma unroll
            for (int kc = 0; kc < 8; ++kc) {
                v16bf a = load_a_frag(h1s, HID, kc * 32, lane);
                c0 = __builtin_amdgcn_wmma_f32_16x16x32_bf16(
                        false, a, false, w2f[0][kc], (short)0, c0, false, false);
                c1 = __builtin_amdgcn_wmma_f32_16x16x32_bf16(
                        false, a, false, w2f[1][kc], (short)0, c1, false, false);
            }
            store_silu(h2s, c0, wave * 2 + 0);
            store_silu(h2s, c1, wave * 2 + 1);
        }
        __syncthreads();
        // layer 3: h2(16x256) @ W3(256x7); K split across the 8 waves
        {
            v16bf a = load_a_frag(h2s, HID, wave * 32, lane);
            v8f c;
            #pragma unroll
            for (int v = 0; v < 8; ++v) c[v] = 0.0f;
            c = __builtin_amdgcn_wmma_f32_16x16x32_bf16(
                    false, a, false, w3f, (short)0, c, false, false);
            int n = lane & 15, mh = (lane >> 4) << 3;
            #pragma unroll
            for (int v = 0; v < 8; ++v)
                part[wave * 256 + (v + mh) * 16 + n] = c[v];
        }
        __syncthreads();
        if (owner) {
            float s = b3v;
            #pragma unroll
            for (int w = 0; w < 8; ++w) s += part[w * 256 + om * 16 + oc];
            anew = s;
        }
    };

    // ---- init: state registers + first accel at t=0 (tau idx 0) ----
    float q_r = 0.0f, v_r = 0.0f, a_r = 0.0f, qn_r = 0.0f;
    if (owner) {
        q_r = init_state[(size_t)(r0 + om) * SD + oc];
        v_r = init_state[(size_t)(r0 + om) * SD + NDOF + oc];
        xs[om * KPAD + oc]        = (__bf16)q_r;
        xs[om * KPAD + NDOF + oc] = (__bf16)v_r;
        xs[om * KPAD + SD + oc]   = (__bf16)torq[((size_t)(r0 + om) * SEQ + 0) * NDOF + oc];
    }
    if (tid < ROWS * (KPAD - INDIM)) {   // zero the pad columns once
        int m = tid / (KPAD - INDIM), c = INDIM + tid % (KPAD - INDIM);
        xs[m * KPAD + c] = (__bf16)0.0f;
    }
    __syncthreads();
    accel_mlp();
    if (owner) a_r = anew;

    // ---- 256 sequential Verlet steps ----
    for (int i = 0; i < SEQ; ++i) {
        int idx = (i + 1 < SEQ) ? (i + 1) : (SEQ - 1);
        if (owner) {
            float qq = q_r + v_r * DT_F + 0.5f * a_r * DT_F * DT_F;
            float vv = v_r + a_r * DT_F;                  // v_pred
            qn_r = qq;
            xs[om * KPAD + oc]        = (__bf16)qq;
            xs[om * KPAD + NDOF + oc] = (__bf16)vv;
            xs[om * KPAD + SD + oc]   = (__bf16)torq[((size_t)(r0 + om) * SEQ + idx) * NDOF + oc];
            // prefetch next step's torque slice (global_prefetch_b8)
            int idx2 = (i + 2 < SEQ) ? (i + 2) : (SEQ - 1);
            __builtin_prefetch(&torq[((size_t)(r0 + om) * SEQ + idx2) * NDOF + oc], 0, 1);
        }
        __syncthreads();
        accel_mlp();
        if (owner) {
            float vnew = v_r + 0.5f * (a_r + anew) * DT_F;
            q_r = qn_r; v_r = vnew; a_r = anew;
            float* o = out + ((size_t)(r0 + om) * SEQ + i) * SD;
            o[oc]        = qn_r;
            o[NDOF + oc] = vnew;
        }
        // no barrier needed here: next iteration's xs writes are by the same
        // owner threads, and non-owner waves wait at the pre-accel barrier.
    }
}

extern "C" void kernel_launch(void* const* d_in, const int* in_sizes, int n_in,
                              void* d_out, int out_size, void* d_ws, size_t ws_size,
                              hipStream_t stream) {
    const float* init_state = (const float*)d_in[0];
    const float* torq       = (const float*)d_in[1];
    const float* W1         = (const float*)d_in[2];
    const float* b1         = (const float*)d_in[3];
    const float* W2         = (const float*)d_in[4];
    const float* b2         = (const float*)d_in[5];
    const float* W3         = (const float*)d_in[6];
    const float* b3         = (const float*)d_in[7];
    float* out = (float*)d_out;

    dim3 grid(B_TOT / ROWS);   // 256 blocks, one per 16-row batch tile
    dim3 block(256);           // 8 wave32s
    hipLaunchKernelGGL(node_verlet_kernel, grid, block, 0, stream,
                       init_state, torq, W1, b1, W2, b2, W3, b3, out);
}
